// Bottleneck_2001454760192
// MI455X (gfx1250) — matrix operand order repaired and checked by reading, not compiled, not run on hardware
//
#include <hip/hip_runtime.h>

// ---------------------------------------------------------------------------
// Sparse bottleneck block for MI455X (gfx1250, wave32, WMMA).
// All GEMMs run as v_wmma_f32_16x16x32_bf16 with f32 accumulation.
// Register-blocked: each wave owns a 2x4 grid of 16x16 output tiles.
// A-tiles are gather-staged into LDS with GLOBAL_LOAD_ASYNC_TO_LDS_B128
// (ASYNCcnt-tracked DMA), synchronized with s_wait_asynccnt + barrier.
// ---------------------------------------------------------------------------

typedef __bf16 bf16_t;
typedef __attribute__((ext_vector_type(16))) __bf16 v16bf;
typedef __attribute__((ext_vector_type(8)))  float  v8f;

#define N_IN_  200000
#define N_OUT_ 50000
#define M_     60000
#define C_     256
#define K_     27
#define EPS_   1e-5f
#define SLOPE_ 0.01f

__device__ __forceinline__ bf16_t f2bf(float f) {
  union { float f; unsigned u; } v; v.f = f;
  unsigned r = v.u + 0x7fffu + ((v.u >> 16) & 1u);   // round-to-nearest-even
  unsigned short h = (unsigned short)(r >> 16);
  union { unsigned short s; bf16_t b; } o; o.s = h;
  return o.b;
}

// --------------------------- utility kernels -------------------------------

__global__ void zero_f32(float* __restrict__ p, int n) {
  int i = blockIdx.x * blockDim.x + threadIdx.x;
  int s = gridDim.x * blockDim.x;
  for (; i < n; i += s) p[i] = 0.0f;
}

// f32 -> bf16 elementwise
__global__ void f32_to_bf16(const float* __restrict__ in, bf16_t* __restrict__ out, int n) {
  int i = blockIdx.x * blockDim.x + threadIdx.x;
  int s = gridDim.x * blockDim.x;
  for (; i < n; i += s) out[i] = f2bf(in[i]);
}

// Transpose each 256x256 matrix and convert to bf16: out[m][col][k] = in[m][k][col]
__global__ void convert_w_t(const float* __restrict__ in, bf16_t* __restrict__ out, int n) {
  int i = blockIdx.x * blockDim.x + threadIdx.x;
  int s = gridDim.x * blockDim.x;
  for (; i < n; i += s) {
    int m = i >> 16;
    int e = i & 65535;          // e = col*256 + k
    int col = e >> 8;
    int k   = e & 255;
    out[(size_t)m * 65536 + e] = f2bf(in[(size_t)m * 65536 + (size_t)k * 256 + col]);
  }
}

// ------------------------------- WMMA GEMM ---------------------------------
// C[row,:] (+)= A[gather(row), :] @ B   with B pre-transposed: Bt[col][k].
// Block = 256 threads = 8 wave32s arranged 2(M) x 4(N).
// Each wave computes a 2x4 grid of 16x16 tiles -> block tile = 64 rows x 256 cols.
// blockIdx.x = 64-row tile, blockIdx.z = matrix index (rulebook offset k).
__global__ __launch_bounds__(256) void wmma_gemm(
    const bf16_t* __restrict__ A,
    const int* __restrict__ gather, int gStride,
    const bf16_t* __restrict__ Bt, int bStride,
    float* __restrict__ Cout,
    const int* __restrict__ scatter, int sStride,
    int nrows)
{
  __shared__ __align__(16) bf16_t As[64 * 256];   // 32 KB A tile (64 rows x K=256)

  const int t    = threadIdx.x;
  const int z    = blockIdx.z;
  const int row0 = blockIdx.x * 64;

  const bf16_t* Bz = Bt + (size_t)z * bStride;
  const int*    gz = gather  ? gather  + (size_t)z * gStride : nullptr;
  const int*    sz = scatter ? scatter + (size_t)z * sStride : nullptr;

  // Gather-stage the A tile (64 rows x 256 bf16) straight into LDS with the
  // CDNA5 async memory->LDS DMA path (128 B per thread, contiguous).
  {
    int r = t >> 2;            // 0..63 : tile row
    int q = (t & 3) * 64;      // element offset within row (64 bf16 = 8 x 16B)
    int grow = row0 + r;
    unsigned lds_addr = (unsigned)(uintptr_t)&As[r * 256 + q];
    if (grow < nrows) {
      int ri = gz ? gz[grow] : grow;
      const bf16_t* src = A + (size_t)ri * 256 + q;
      #pragma unroll
      for (int j = 0; j < 8; ++j) {
        asm volatile("global_load_async_to_lds_b128 %0, %1, off"
                     :: "v"(lds_addr + 16u * j), "v"(src + 8 * j)
                     : "memory");
      }
    } else {
      uint4 zv = make_uint4(0u, 0u, 0u, 0u);
      uint4* dst = reinterpret_cast<uint4*>(&As[r * 256 + q]);
      #pragma unroll
      for (int j = 0; j < 8; ++j) dst[j] = zv;
    }
  }
  asm volatile("s_wait_asynccnt 0x0" ::: "memory");
  __syncthreads();

  const int wave  = t >> 5;          // 0..7
  const int lane  = t & 31;
  const int waveM = wave >> 2;       // 0..1
  const int waveN = wave & 3;        // 0..3
  const int mrow0 = waveM * 32;      // two M-tiles: mrow0, mrow0+16
  const int ncol0 = waveN * 64;      // four N-tiles: ncol0 + 16*j

  const int m     = lane & 15;       // A row owned by this lane
  const int khalf = lane >> 4;       // A K-half select (ISA 16-bit A layout)
  const int nloc  = lane & 15;       // B/C column owned by this lane
  const int kb16  = khalf * 8;       // B K-base for lane half

  v8f acc[2][4] = {};

  const bf16_t* a0base = &As[(mrow0 + m) * 256 + khalf * 8];
  const bf16_t* a1base = a0base + 16 * 256;

  #pragma unroll
  for (int k0 = 0; k0 < 256; k0 += 32) {
    union Frag { uint4 q[2]; v16bf v; };
    Frag a0, a1, b[4];
    // A 16x32 bf16 fragments: lanes 0-15 K in {0..7,16..23}, lanes 16-31 {8..15,24..31}
    a0.q[0] = *reinterpret_cast<const uint4*>(a0base + k0);
    a0.q[1] = *reinterpret_cast<const uint4*>(a0base + k0 + 16);
    a1.q[0] = *reinterpret_cast<const uint4*>(a1base + k0);
    a1.q[1] = *reinterpret_cast<const uint4*>(a1base + k0 + 16);
    // B 32x16 bf16 fragments: same K split as A (transposed weights)
    #pragma unroll
    for (int j = 0; j < 4; ++j) {
      const bf16_t* Brow = Bz + (size_t)(ncol0 + 16 * j + nloc) * 256 + k0 + kb16;
      b[j].q[0] = *reinterpret_cast<const uint4*>(Brow);
      b[j].q[1] = *reinterpret_cast<const uint4*>(Brow + 16);
    }
    #pragma unroll
    for (int j = 0; j < 4; ++j) {
      acc[0][j] = __builtin_amdgcn_wmma_f32_16x16x32_bf16(false, a0.v, false, b[j].v,
                                                          (short)0, acc[0][j], false, false);
      acc[1][j] = __builtin_amdgcn_wmma_f32_16x16x32_bf16(false, a1.v, false, b[j].v,
                                                          (short)0, acc[1][j], false, false);
    }
  }

  // C/D layout: VGPR i -> M = i + 8*(lane>=16), N = lane&15
  const int rbase = khalf * 8;
  if (row0 + 64 <= nrows) {
    // Full tile: no per-row bounds checks.
    #pragma unroll
    for (int mt = 0; mt < 2; ++mt) {
      #pragma unroll
      for (int j = 0; j < 4; ++j) {
        const int col = ncol0 + 16 * j + nloc;
        #pragma unroll
        for (int i = 0; i < 8; ++i) {
          int grow = row0 + mrow0 + mt * 16 + rbase + i;
          if (sz) atomicAdd(&Cout[(size_t)sz[grow] * 256 + col], acc[mt][j][i]);
          else    Cout[(size_t)grow * 256 + col] = acc[mt][j][i];
        }
      }
    }
  } else {
    #pragma unroll
    for (int mt = 0; mt < 2; ++mt) {
      #pragma unroll
      for (int j = 0; j < 4; ++j) {
        const int col = ncol0 + 16 * j + nloc;
        #pragma unroll
        for (int i = 0; i < 8; ++i) {
          int grow = row0 + mrow0 + mt * 16 + rbase + i;
          if (grow < nrows) {
            if (sz) atomicAdd(&Cout[(size_t)sz[grow] * 256 + col], acc[mt][j][i]);
            else    Cout[(size_t)grow * 256 + col] = acc[mt][j][i];
          }
        }
      }
    }
  }
}

// --------------------------- GroupNorm kernels -----------------------------
// stats: 16 floats per set = {sum, sumsq} x 8 groups (channels c: group = c>>5).
// stride is a multiple of 256, so each thread's channel (hence group) is fixed.
__global__ void gn_stats(const float* __restrict__ x, int n, float* __restrict__ stats) {
  int tid = blockIdx.x * blockDim.x + threadIdx.x;
  int s   = gridDim.x * blockDim.x;
  int g   = (tid >> 5) & 7;
  float sum = 0.f, ss = 0.f;
  for (int i = tid; i < n; i += s) {
    float v = x[i];
    sum += v; ss += v * v;
  }
  atomicAdd(&stats[g * 2 + 0], sum);
  atomicAdd(&stats[g * 2 + 1], ss);
}

__global__ void gn_apply_bf16(const float* __restrict__ in, const float* __restrict__ stats,
                              const float* __restrict__ gamma, const float* __restrict__ beta,
                              float inv_cnt, int do_lrelu,
                              bf16_t* __restrict__ out, int n) {
  int i = blockIdx.x * blockDim.x + threadIdx.x;
  int s = gridDim.x * blockDim.x;
  for (; i < n; i += s) {
    int c = i & 255, g = c >> 5;
    float mean = stats[g * 2] * inv_cnt;
    float var  = stats[g * 2 + 1] * inv_cnt - mean * mean;
    float sc   = rsqrtf(var + EPS_);
    float v = (in[i] - mean) * sc * gamma[c] + beta[c];
    if (do_lrelu && v < 0.f) v *= SLOPE_;
    out[i] = f2bf(v);
  }
}

__global__ void final_combine(const float* __restrict__ y2, const float* __restrict__ ds,
                              const float* __restrict__ st3, const float* __restrict__ std_,
                              const float* __restrict__ g3, const float* __restrict__ b3,
                              const float* __restrict__ gd, const float* __restrict__ bd,
                              float inv3, float invd, float* __restrict__ out, int n) {
  int i = blockIdx.x * blockDim.x + threadIdx.x;
  int s = gridDim.x * blockDim.x;
  for (; i < n; i += s) {
    int c = i & 255, g = c >> 5;
    float m3 = st3[g * 2] * inv3;
    float v3 = st3[g * 2 + 1] * inv3 - m3 * m3;
    float a  = (y2[i] - m3) * rsqrtf(v3 + EPS_) * g3[c] + b3[c];
    float md = std_[g * 2] * invd;
    float vd = std_[g * 2 + 1] * invd - md * md;
    float b  = (ds[i] - md) * rsqrtf(vd + EPS_) * gd[c] + bd[c];
    out[i] = a + b;
  }
}

// ------------------------------- launcher ----------------------------------

extern "C" void kernel_launch(void* const* d_in, const int* in_sizes, int n_in,
                              void* d_out, int out_size, void* d_ws, size_t ws_size,
                              hipStream_t stream) {
  (void)in_sizes; (void)n_in; (void)out_size; (void)ws_size;

  const float* x    = (const float*)d_in[0];
  const float* W1a  = (const float*)d_in[1];
  const float* g1   = (const float*)d_in[2];
  const float* b1   = (const float*)d_in[3];
  const float* W3   = (const float*)d_in[4];
  const float* g2   = (const float*)d_in[5];
  const float* b2   = (const float*)d_in[6];
  const float* W1b  = (const float*)d_in[7];
  const float* g3   = (const float*)d_in[8];
  const float* b3   = (const float*)d_in[9];
  const float* Wd   = (const float*)d_in[10];
  const float* gd   = (const float*)d_in[11];
  const float* bd   = (const float*)d_in[12];
  const int* in_idx  = (const int*)d_in[13];
  const int* out_idx = (const int*)d_in[14];
  const int* ds_idx  = (const int*)d_in[15];
  float* out = (float*)d_out;

  // Workspace layout (16B aligned offsets).
  char* ws = (char*)d_ws;
  bf16_t* x_bf   = (bf16_t*)(ws);                                  // 102,400,000 B
  bf16_t* h_bf   = (bf16_t*)(ws + 102400000u);                     // 102,400,000 B
  bf16_t* ygn_bf = (bf16_t*)(ws + 204800000u);                     //  25,600,000 B
  bf16_t* Wt     = (bf16_t*)(ws + 230400000u);                     //   3,932,160 B (30 matrices)
  float*  h0     = (float*) (ws + 234332160u);                     // 204,800,000 B
  float*  y      = h0;                                             // alias: h0 dead after GN1
  float*  y2     = (float*) (ws + 234332160u + 51200000u);
  float*  ds0    = (float*) (ws + 234332160u + 102400000u);
  float*  stats  = (float*) (ws + 234332160u + 204800000u);        // 4 sets x 16 floats

  bf16_t* W1a_t = Wt;
  bf16_t* W3_t  = Wt + 1 * 65536;
  bf16_t* W1b_t = Wt + 28 * 65536;
  bf16_t* Wd_t  = Wt + 29 * 65536;

  const float inv_in  = 1.0f / ((float)N_IN_  * 32.0f);
  const float inv_out = 1.0f / ((float)N_OUT_ * 32.0f);

  // Deterministic zeroing each call (y is atomic-accumulated, stats atomic-reduced).
  zero_f32<<<1, 64, 0, stream>>>(stats, 64);
  zero_f32<<<4096, 256, 0, stream>>>(y, N_OUT_ * C_);

  // Weight transpose + bf16 conversion; x -> bf16.
  convert_w_t<<<256,  256, 0, stream>>>(W1a, W1a_t, 1 * 65536);
  convert_w_t<<<2048, 256, 0, stream>>>(W3,  W3_t, 27 * 65536);
  convert_w_t<<<256,  256, 0, stream>>>(W1b, W1b_t, 1 * 65536);
  convert_w_t<<<256,  256, 0, stream>>>(Wd,  Wd_t,  1 * 65536);
  f32_to_bf16<<<4096, 256, 0, stream>>>(x, x_bf, N_IN_ * C_);

  // h0 = x @ W1a
  dim3 gA((N_IN_ + 63) / 64, 1, 1);
  wmma_gemm<<<gA, 256, 0, stream>>>(x_bf, nullptr, 0, W1a_t, 0, h0, nullptr, 0, N_IN_);
  gn_stats<<<1024, 256, 0, stream>>>(h0, N_IN_ * C_, stats + 0);
  gn_apply_bf16<<<4096, 256, 0, stream>>>(h0, stats + 0, g1, b1, inv_in, 1, h_bf, N_IN_ * C_);

  // y[out_idx[k]] += h[in_idx[k]] @ W3[k], k = 0..26 via gridDim.z
  dim3 gC((M_ + 63) / 64, 1, K_);
  wmma_gemm<<<gC, 256, 0, stream>>>(h_bf, in_idx, M_, W3_t, 65536, y, out_idx, M_, M_);
  gn_stats<<<1024, 256, 0, stream>>>(y, N_OUT_ * C_, stats + 16);
  gn_apply_bf16<<<2048, 256, 0, stream>>>(y, stats + 16, g2, b2, inv_out, 0, ygn_bf, N_OUT_ * C_);

  // y2 = gn2(y) @ W1b
  dim3 gB((N_OUT_ + 63) / 64, 1, 1);
  wmma_gemm<<<gB, 256, 0, stream>>>(ygn_bf, nullptr, 0, W1b_t, 0, y2, nullptr, 0, N_OUT_);
  gn_stats<<<1024, 256, 0, stream>>>(y2, N_OUT_ * C_, stats + 32);

  // ds0 = x[ds_idx] @ Wd
  wmma_gemm<<<gB, 256, 0, stream>>>(x_bf, ds_idx, 0, Wd_t, 0, ds0, nullptr, 0, N_OUT_);
  gn_stats<<<1024, 256, 0, stream>>>(ds0, N_OUT_ * C_, stats + 48);

  // out = gn3(y2) + gnd(ds0)
  final_combine<<<2048, 256, 0, stream>>>(y2, ds0, stats + 32, stats + 48,
                                          g3, b3, gd, bd, inv_out, inv_out,
                                          out, N_OUT_ * C_);
}
